// CovLayer_80693845557391
// MI455X (gfx1250) — matrix operand order repaired and checked by reading, not compiled, not run on hardware
//
#include <hip/hip_runtime.h>
#include <hip/hip_bf16.h>

typedef __attribute__((ext_vector_type(8)))  float    v8f;
typedef __attribute__((ext_vector_type(2)))  _Float16 h2;
typedef __attribute__((ext_vector_type(2)))  __fp16   fp16x2;
typedef __attribute__((ext_vector_type(4)))  _Float16 h4;
typedef __attribute__((ext_vector_type(8)))  _Float16 v8h;
typedef __attribute__((ext_vector_type(16))) _Float16 v16h;

#define B_DIM 64
#define C_DIM 256
#define T_DIM 2048
#define KC    32            // K per LDS chunk == one WMMA K-step
#define LDSS  40            // row stride in f16: 80B = 20 banks -> conflict-free

// ---------------------------------------------------------------------------
// Kernel 1: per-row means  m[b,c] = (1/T) * sum_t X[b,c,t]
// ---------------------------------------------------------------------------
__global__ __launch_bounds__(256) void cov_row_means(const float* __restrict__ X,
                                                     float* __restrict__ means) {
    const int warp = threadIdx.x >> 5;
    const int lane = threadIdx.x & 31;
    const int row  = blockIdx.x * 8 + warp;
    const float4* xr = (const float4*)(X + (size_t)row * T_DIM);
    float s = 0.0f;
#pragma unroll 4
    for (int i = lane; i < T_DIM / 4; i += 32) {
        float4 v = xr[i];
        s += v.x + v.y + v.z + v.w;
    }
#pragma unroll
    for (int off = 16; off > 0; off >>= 1)
        s += __shfl_xor(s, off, 32);
    if (lane == 0)
        means[row] = s * (1.0f / (float)T_DIM);
}

// ---------------------------------------------------------------------------
// fp32 -> fp16 hi/lo split using v_cvt_pkrtz_f16_f32 (~3 VALU/elem)
// ---------------------------------------------------------------------------
static __device__ __forceinline__ h2 pk_f16(float a, float b) {
    fp16x2 t = __builtin_amdgcn_cvt_pkrtz(a, b);
    return __builtin_bit_cast(h2, t);
}
static __device__ __forceinline__ void split4(float4 v, h4& hh, h4& ll) {
    h2 h01 = pk_f16(v.x, v.y);
    h2 h23 = pk_f16(v.z, v.w);
    float r0 = v.x - (float)h01[0];
    float r1 = v.y - (float)h01[1];
    float r2 = v.z - (float)h23[0];
    float r3 = v.w - (float)h23[1];
    h2 l01 = pk_f16(r0, r1);
    h2 l23 = pk_f16(r2, r3);
    hh = __builtin_shufflevector(h01, h23, 0, 1, 2, 3);
    ll = __builtin_shufflevector(l01, l23, 0, 1, 2, 3);
}

// operand assembly from LDS (f16 units)
// A 16x32 layout: lane(m,hi): K = {8*hi .. +7} and {16+8*hi .. +7}
// B 32x16 layout: lane(n,hi): K = {8*hi .. +7} and {16+8*hi .. +7}
static __device__ __forceinline__ v16h load_opA(const _Float16* p, int row, int hi) {
    const _Float16* q = p + row * LDSS + 8 * hi;
    v8h x = *(const v8h*)(q);
    v8h y = *(const v8h*)(q + 16);
    return __builtin_shufflevector(x, y, 0,1,2,3,4,5,6,7,8,9,10,11,12,13,14,15);
}
static __device__ __forceinline__ v16h load_opB(const _Float16* p, int row, int hi) {
    const _Float16* q = p + row * LDSS + 8 * hi;
    v8h x = *(const v8h*)(q);
    v8h y = *(const v8h*)(q + 16);
    return __builtin_shufflevector(x, y, 0,1,2,3,4,5,6,7,8,9,10,11,12,13,14,15);
}

#define WMMA_F16(A, Bv, Cacc) \
    __builtin_amdgcn_wmma_f32_16x16x32_f16(false, (A), false, (Bv), (short)0, (Cacc), false, false)

// ---------------------------------------------------------------------------
// Kernel 2: cov[b] = (X X^T)/T - m m^T via 3xFP16-split WMMA (fp32-grade)
// workgroup: 128x128 output block, 8 waves, each wave owns 32x64 (2x4 tiles)
// software-pipelined: prefetch chunk k+1 into VGPRs while chunk k runs on XDL
// ---------------------------------------------------------------------------
__global__ __launch_bounds__(256) void cov_wmma_f16x3(const float* __restrict__ X,
                                                      const float* __restrict__ means,
                                                      float* __restrict__ out) {
    // 4 strips (A hi/lo, B hi/lo) of 128 rows x 40 f16 = 40 KB
    __shared__ __align__(16) _Float16 smem[4 * 128 * LDSS];
    _Float16* AH = smem;
    _Float16* AL = AH + 128 * LDSS;
    _Float16* BH = AL + 128 * LDSS;
    _Float16* BL = BH + 128 * LDSS;

    const int b  = blockIdx.y;
    const int i0 = (blockIdx.x >> 1) * 128;     // 2 row-blocks
    const int j0 = (blockIdx.x & 1) * 128;      // 2 col-blocks
    const bool diag = (i0 == j0);               // B strip == A strip

    const int tid  = threadIdx.x;
    const int wid  = tid >> 5;
    const int lane = tid & 31;
    const int m    = lane & 15;
    const int hi   = lane >> 4;

    const int wi = wid & 3;                     // 4 row-waves: 32 rows each
    const int wj = wid >> 2;                    // 2 col-waves: 64 cols each

    // B operand source: reuse A strips on diagonal blocks
    const _Float16* BHs = diag ? AH : BH;
    const _Float16* BLs = diag ? AL : BL;

    const float* Xb = X + (size_t)b * C_DIM * T_DIM;

    // staging geometry: thread covers rows r0 + 32*it, fixed column group c4
    const int r0 = tid >> 3;                    // 0..31
    const int c4 = (tid & 7) << 2;              // 0,4,...,28
    const float* gA = Xb + (size_t)(i0 + r0) * T_DIM + c4;
    const float* gB = Xb + (size_t)(j0 + r0) * T_DIM + c4;

    v8f acc[2][4] = {};
    float4 ra[4], rb[4];

    // ---- prologue: load + stage chunk 0 ----
#pragma unroll
    for (int it = 0; it < 4; ++it)
        ra[it] = *(const float4*)(gA + (size_t)(32 * it) * T_DIM);
    if (!diag) {
#pragma unroll
        for (int it = 0; it < 4; ++it)
            rb[it] = *(const float4*)(gB + (size_t)(32 * it) * T_DIM);
    }
#pragma unroll
    for (int it = 0; it < 4; ++it) {
        const int row = r0 + 32 * it;
        h4 hh, ll;
        split4(ra[it], hh, ll);
        *(h4*)&AH[row * LDSS + c4] = hh;
        *(h4*)&AL[row * LDSS + c4] = ll;
        if (!diag) {
            split4(rb[it], hh, ll);
            *(h4*)&BH[row * LDSS + c4] = hh;
            *(h4*)&BL[row * LDSS + c4] = ll;
        }
    }
    __syncthreads();

    for (int kk = 0; kk < T_DIM; kk += KC) {
        // ---- issue next chunk's global loads (latency hides under WMMA) ----
        const int kn = (kk + KC < T_DIM) ? kk + KC : 0;   // wrap: harmless restage
#pragma unroll
        for (int it = 0; it < 4; ++it)
            ra[it] = *(const float4*)(gA + (size_t)(32 * it) * T_DIM + kn);
        if (!diag) {
#pragma unroll
            for (int it = 0; it < 4; ++it)
                rb[it] = *(const float4*)(gB + (size_t)(32 * it) * T_DIM + kn);
        }

        // ---- one K-step of 32: 24 f16 WMMAs per wave ----
        v16h aH0 = load_opA(AH, wi * 32 + m,      hi);
        v16h aH1 = load_opA(AH, wi * 32 + 16 + m, hi);
        v16h aL0 = load_opA(AL, wi * 32 + m,      hi);
        v16h aL1 = load_opA(AL, wi * 32 + 16 + m, hi);
#pragma unroll
        for (int y = 0; y < 4; ++y) {
            const int rB = wj * 64 + y * 16 + m;
            v16h bH = load_opB(BHs, rB, hi);
            v16h bL = load_opB(BLs, rB, hi);
            acc[0][y] = WMMA_F16(aH0, bH, acc[0][y]);
            acc[0][y] = WMMA_F16(aH0, bL, acc[0][y]);
            acc[0][y] = WMMA_F16(aL0, bH, acc[0][y]);
            acc[1][y] = WMMA_F16(aH1, bH, acc[1][y]);
            acc[1][y] = WMMA_F16(aH1, bL, acc[1][y]);
            acc[1][y] = WMMA_F16(aL1, bH, acc[1][y]);
        }

        __syncthreads();                        // all waves done reading chunk k
        // ---- convert + store chunk k+1 ----
#pragma unroll
        for (int it = 0; it < 4; ++it) {
            const int row = r0 + 32 * it;
            h4 hh, ll;
            split4(ra[it], hh, ll);
            *(h4*)&AH[row * LDSS + c4] = hh;
            *(h4*)&AL[row * LDSS + c4] = ll;
            if (!diag) {
                split4(rb[it], hh, ll);
                *(h4*)&BH[row * LDSS + c4] = hh;
                *(h4*)&BL[row * LDSS + c4] = ll;
            }
        }
        __syncthreads();                        // chunk k+1 visible
    }

    // ---- epilogue: out = acc/T - m_row * m_col ----
    const float invT = 1.0f / (float)T_DIM;
    const float* mb  = means + b * C_DIM;
    float* outb = out + (size_t)b * C_DIM * C_DIM;

#pragma unroll
    for (int y = 0; y < 4; ++y) {
        const int col = j0 + wj * 64 + y * 16 + m;
        const float mc = mb[col];
#pragma unroll
        for (int x = 0; x < 2; ++x) {
#pragma unroll
            for (int r = 0; r < 8; ++r) {
                const int row = i0 + wi * 32 + x * 16 + r + 8 * hi;  // M = r + 8*hi
                outb[(size_t)row * C_DIM + col] = acc[x][y][r] * invT - mb[row] * mc;
            }
        }
    }
}

// ---------------------------------------------------------------------------
extern "C" void kernel_launch(void* const* d_in, const int* in_sizes, int n_in,
                              void* d_out, int out_size, void* d_ws, size_t ws_size,
                              hipStream_t stream) {
    const float* X = (const float*)d_in[0];
    float* out     = (float*)d_out;
    float* means   = (float*)d_ws;                  // B*C floats = 64 KB scratch

    cov_row_means<<<(B_DIM * C_DIM) / 8, 256, 0, stream>>>(X, means);

    dim3 grid(4, B_DIM);                            // 2 x 2 blocks of 128x128 per batch
    cov_wmma_f16x3<<<grid, 256, 0, stream>>>(X, means, out);
}
